// Decoder_52183852647094
// MI455X (gfx1250) — matrix operand order repaired and checked by reading, not compiled, not run on hardware
//
#include <hip/hip_runtime.h>
#include <cstdint>
#include <cstddef>

typedef __attribute__((ext_vector_type(16))) _Float16 v16h;
typedef __attribute__((ext_vector_type(8)))  float    v8f;

#define B_   64
#define E_   256
#define Z_   256
#define H_   512
#define V_   32000
#define T_   64
#define G4H  2048          // 4*H
#define K1   1024          // concat [x(512); h(512)]
#define KT1  32            // K1 / 32
#define KT2  16            // H  / 32
#define MT_  4             // B_/16 M-tiles, register-blocked into one wave

// ---------------------------------------------------------------------------
// WMMA operand swizzles (CDNA5 ISA 7.12.2, wave32, 16-bit elements)
// A tile 16(M)x32(K): lane l holds row M=l%16; half j -> K=(j%8)+16*(j/8)+8*(l/16)
// B tile 32(K)x16(N): lane l holds col N=l%16; half j -> K=(j%8)+16*(j/8)+8*(l/16)
// Tiles stored contiguously as [lane][16 halves] so each lane loads one v16h.
// ---------------------------------------------------------------------------
__device__ __forceinline__ int a_swz(int m, int k, int ktiles) {
  int kk = k & 31;
  int l  = (m & 15) | (((kk >> 3) & 1) << 4);
  int j  = (kk & 7) | ((kk >> 4) << 3);
  return ((((m >> 4) * ktiles) + (k >> 5)) << 9) | (l << 4) | j;
}
__device__ __forceinline__ int b_swz(int n, int k, int ktiles) {
  int kk = k & 31;
  int l  = (n & 15) | (((kk >> 3) & 1) << 4);
  int j  = (kk & 7) | ((kk >> 4) << 3);
  return ((((n >> 4) * ktiles) + (k >> 5)) << 9) | (l << 4) | j;
}

__device__ __forceinline__ float sigm(float x) { return 1.0f / (1.0f + expf(-x)); }

// counter-based hash -> uniform (0,1) for Gumbel sampling (stand-in for threefry)
__device__ __forceinline__ float hash_uniform(unsigned t, unsigned b, unsigned v) {
  unsigned h = v * 2654435761u + b * 2246822519u + t * 3266489917u + 0x9E3779B9u;
  h ^= h >> 16; h *= 0x7feb352du; h ^= h >> 15; h *= 0x846ca68bu; h ^= h >> 16;
  return ((h >> 8) + 0.5f) * (1.0f / 16777216.0f);
}

// ---------------------------------------------------------------------------
// prep kernels (run every call; deterministic)
// ---------------------------------------------------------------------------
__global__ void zero_ws(uint32_t* p, int n) {
  int i = blockIdx.x * blockDim.x + threadIdx.x;
  if (i < n) p[i] = 0u;
}

__global__ void f32_to_f16(const float* __restrict__ s, _Float16* __restrict__ d, int n) {
  int i = blockIdx.x * blockDim.x + threadIdx.x;
  if (i < n) d[i] = (_Float16)s[i];
}

// concat [W_ih | W_hh] ([4H x 512] each) -> B-swizzled fp16 [4H x 1024]
__global__ void swz_b_cat(const float* __restrict__ Wih, const float* __restrict__ Whh,
                          _Float16* __restrict__ dst) {
  int i = blockIdx.x * blockDim.x + threadIdx.x;
  if (i >= G4H * K1) return;
  int n = i >> 10, k = i & (K1 - 1);
  float v = (k < H_) ? Wih[n * H_ + k] : Whh[n * H_ + (k - H_)];
  dst[b_swz(n, k, KT1)] = (_Float16)v;
}

// Wout [V x H] row-major -> B-swizzled fp16
__global__ void swz_b_out(const float* __restrict__ W, _Float16* __restrict__ dst, int n_elems) {
  int i = blockIdx.x * blockDim.x + threadIdx.x;
  if (i >= n_elems) return;
  int n = i >> 9, k = i & (H_ - 1);
  dst[b_swz(n, k, KT2)] = (_Float16)W[i];
}

// ---------------------------------------------------------------------------
// per-step kernels
// ---------------------------------------------------------------------------
// fill A1 cols [0,512): concat(emb[s_prev], z), A-swizzled fp16
__global__ void build_x1(const int* __restrict__ sprev, const _Float16* __restrict__ emb16,
                         const float* __restrict__ z, _Float16* __restrict__ A1) {
  int i = blockIdx.x * blockDim.x + threadIdx.x;   // 64*512
  int b = i >> 9, k = i & 511;
  _Float16 v = (k < E_) ? emb16[(size_t)sprev[b] * E_ + k]
                        : (_Float16)z[b * Z_ + (k - E_)];
  A1[a_swz(b, k, KT1)] = v;
}

// WMMA GEMM, M register-blocked: one wave owns one 16-wide N tile and ALL
// 4 M tiles (M=64). Each B (weight) tile is loaded exactly once per wave,
// quartering L2 weight traffic vs one-tile-per-wave.
__global__ __launch_bounds__(256) void gemm16(const _Float16* __restrict__ A,
                                              const _Float16* __restrict__ Bm,
                                              const float* __restrict__ bias,
                                              float* __restrict__ C,
                                              int Ntiles, int Ktiles, int N) {
  int wave = blockIdx.x * 8 + (threadIdx.x >> 5);
  int lane = threadIdx.x & 31;
  if (wave >= Ntiles) return;                      // per-wave uniform: EXEC all-ones
  int nt = wave;
  const _Float16* a0 = A  + (lane << 4);
  const _Float16* bp = Bm + (((size_t)nt * Ktiles) << 9) + (lane << 4);
  v8f acc0 = {}, acc1 = {}, acc2 = {}, acc3 = {};
  const size_t mstride = (size_t)Ktiles << 9;      // halves between M tiles
  for (int kt = 0; kt < Ktiles; ++kt) {
    v16h b  = *(const v16h*)bp;  bp += 512;
    const _Float16* ap = a0 + ((size_t)kt << 9);
    v16h aA = *(const v16h*)(ap);
    v16h aB = *(const v16h*)(ap + mstride);
    v16h aC = *(const v16h*)(ap + 2 * mstride);
    v16h aD = *(const v16h*)(ap + 3 * mstride);
    acc0 = __builtin_amdgcn_wmma_f32_16x16x32_f16(false, aA, false, b, (short)0, acc0, false, false);
    acc1 = __builtin_amdgcn_wmma_f32_16x16x32_f16(false, aB, false, b, (short)0, acc1, false, false);
    acc2 = __builtin_amdgcn_wmma_f32_16x16x32_f16(false, aC, false, b, (short)0, acc2, false, false);
    acc3 = __builtin_amdgcn_wmma_f32_16x16x32_f16(false, aD, false, b, (short)0, acc3, false, false);
  }
  int   col   = (nt << 4) + (lane & 15);
  float bv    = bias[col];
  int   rbase = (lane >> 4) << 3;                  // 0 or 8 within a 16-row tile
  float* Cc   = C + col;
#pragma unroll
  for (int r = 0; r < 8; ++r) {
    Cc[(size_t)(rbase + r) * N]      = acc0[r] + bv;
    Cc[(size_t)(16 + rbase + r) * N] = acc1[r] + bv;
    Cc[(size_t)(32 + rbase + r) * N] = acc2[r] + bv;
    Cc[(size_t)(48 + rbase + r) * N] = acc3[r] + bv;
  }
}

// apply LSTM gates (torch order i,f,g,o); write h (fp16) into two A-swizzled dests
__global__ void lstm_gates(const float* __restrict__ G, float* __restrict__ c,
                           _Float16* __restrict__ dA, int aOff, int aKt,
                           _Float16* __restrict__ dB, int bOff, int bKt) {
  int i = blockIdx.x * blockDim.x + threadIdx.x;   // 64*512
  int b = i >> 9, u = i & (H_ - 1);
  const float* g0 = G + (size_t)b * G4H;
  float gi = g0[u], gf = g0[H_ + u], gg = g0[2 * H_ + u], go = g0[3 * H_ + u];
  float cn = sigm(gf) * c[i] + sigm(gi) * tanhf(gg);
  float hn = sigm(go) * tanhf(cn);
  c[i] = cn;
  _Float16 hh = (_Float16)hn;
  dA[a_swz(b, aOff + u, aKt)] = hh;
  dB[a_swz(b, bOff + u, bKt)] = hh;
}

// per-row: softmax stats + Gumbel-argmax sample + probability writeout
__global__ __launch_bounds__(1024) void softmax_sample(const float* __restrict__ logits,
                                                       float* __restrict__ out_s,
                                                       float* __restrict__ out_p,
                                                       int* __restrict__ sprev, int t) {
  __shared__ float red[1024];
  __shared__ float red2[1024];
  __shared__ int   redi[1024];
  int b = blockIdx.x, tid = threadIdx.x;
  const float* L = logits + (size_t)b * V_;

  // pass 1: row max
  float m = -3.4e38f;
  for (int v = tid; v < V_; v += 1024) m = fmaxf(m, L[v]);
  red[tid] = m; __syncthreads();
  for (int off = 512; off > 0; off >>= 1) {
    if (tid < off) red[tid] = fmaxf(red[tid], red[tid + off]);
    __syncthreads();
  }
  float mx = red[0]; __syncthreads();

  // pass 2: sumexp + Gumbel argmax
  float s = 0.0f, best = -3.4e38f; int bi = 0;
  for (int v = tid; v < V_; v += 1024) {
    float l = L[v];
    s += expf(l - mx);
    float u  = hash_uniform((unsigned)t, (unsigned)b, (unsigned)v);
    float sc = l - logf(-logf(u));
    if (sc > best) { best = sc; bi = v; }
  }
  red[tid] = s; red2[tid] = best; redi[tid] = bi; __syncthreads();
  for (int off = 512; off > 0; off >>= 1) {
    if (tid < off) {
      red[tid] += red[tid + off];
      if (red2[tid + off] > red2[tid]) { red2[tid] = red2[tid + off]; redi[tid] = redi[tid + off]; }
    }
    __syncthreads();
  }
  float inv = 1.0f / red[0];
  int   sel = redi[0];

  // pass 3: write probabilities [B,T,V]
  float* P = out_p + ((size_t)b * T_ + t) * V_;
  for (int v = tid; v < V_; v += 1024) P[v] = expf(L[v] - mx) * inv;
  if (tid == 0) { sprev[b] = sel; out_s[(size_t)b * T_ + t] = (float)sel; }
}

// ---------------------------------------------------------------------------
extern "C" void kernel_launch(void* const* d_in, const int* in_sizes, int n_in,
                              void* d_out, int out_size, void* d_ws, size_t ws_size,
                              hipStream_t stream) {
  (void)in_sizes; (void)n_in; (void)out_size; (void)ws_size;
  const float* z    = (const float*)d_in[0];
  // d_in[1] = maxtime (device scalar); hardcoded T_=64 to stay graph-capturable
  const float* embW = (const float*)d_in[2];
  const float* Wih1 = (const float*)d_in[3];
  const float* Whh1 = (const float*)d_in[4];
  const float* b1   = (const float*)d_in[5];
  const float* Wih2 = (const float*)d_in[6];
  const float* Whh2 = (const float*)d_in[7];
  const float* b2   = (const float*)d_in[8];
  const float* Wout = (const float*)d_in[9];
  const float* bout = (const float*)d_in[10];

  char*  ws  = (char*)d_ws;
  size_t off = 0;
  auto carve = [&](size_t bytes) -> char* {
    char* p = ws + off; off = (off + bytes + 1023) & ~(size_t)1023; return p;
  };
  // --- zero-init region (state that must start at 0 each call) ---
  _Float16* A1  = (_Float16*)carve((size_t)B_ * K1 * 2);   // [emb;z | h1] A-swizzled
  _Float16* A2  = (_Float16*)carve((size_t)B_ * K1 * 2);   // [h1 | h2]    A-swizzled
  _Float16* Ah2 = (_Float16*)carve((size_t)B_ * H_ * 2);   // h2           A-swizzled
  float*    c1  = (float*)carve((size_t)B_ * H_ * 4);
  float*    c2  = (float*)carve((size_t)B_ * H_ * 4);
  int*   sprev  = (int*)carve((size_t)B_ * 4);             // START_ID == 0
  size_t zero_bytes = off;
  // --- weights / scratch ---
  _Float16* emb16 = (_Float16*)carve((size_t)V_ * E_ * 2);
  _Float16* W1sw  = (_Float16*)carve((size_t)G4H * K1 * 2);
  _Float16* W2sw  = (_Float16*)carve((size_t)G4H * K1 * 2);
  _Float16* Wosw  = (_Float16*)carve((size_t)V_ * H_ * 2);
  float* gates1   = (float*)carve((size_t)B_ * G4H * 4);
  float* gates2   = (float*)carve((size_t)B_ * G4H * 4);
  float* logits   = (float*)carve((size_t)B_ * V_ * 4);

  float* out_s = (float*)d_out;                 // [B,T] token ids (as float)
  float* out_p = out_s + (size_t)B_ * T_;       // [B,T,V] probabilities

  // ---- one-time (per call) prep ----
  {
    int n = (int)(zero_bytes / 4);
    zero_ws<<<(n + 255) / 256, 256, 0, stream>>>((uint32_t*)d_ws, n);
  }
  {
    int n = V_ * E_;
    f32_to_f16<<<(n + 255) / 256, 256, 0, stream>>>(embW, emb16, n);
  }
  {
    int n = G4H * K1;
    swz_b_cat<<<(n + 255) / 256, 256, 0, stream>>>(Wih1, Whh1, W1sw);
    swz_b_cat<<<(n + 255) / 256, 256, 0, stream>>>(Wih2, Whh2, W2sw);
  }
  {
    int n = V_ * H_;
    swz_b_out<<<(n + 255) / 256, 256, 0, stream>>>(Wout, Wosw, n);
  }

  // ---- autoregressive time loop ----
  for (int t = 0; t < T_; ++t) {
    build_x1<<<128, 256, 0, stream>>>(sprev, emb16, z, A1);
    // LSTM1: gates1 = [x;h1] x [Wih1|Whh1]^T + b1  (Ntiles=128, Ktiles=32; 128 waves)
    gemm16<<<16, 256, 0, stream>>>(A1, W1sw, b1, gates1, G4H / 16, KT1, G4H);
    lstm_gates<<<128, 256, 0, stream>>>(gates1, c1, A2, 0, KT1, A1, H_, KT1);
    // LSTM2: gates2 = [h1;h2] x [Wih2|Whh2]^T + b2
    gemm16<<<16, 256, 0, stream>>>(A2, W2sw, b2, gates2, G4H / 16, KT1, G4H);
    lstm_gates<<<128, 256, 0, stream>>>(gates2, c2, A2, H_, KT1, Ah2, 0, KT2);
    // logits = h2 x Wout^T + bout  (Ntiles=2000, Ktiles=16; 2000 waves)
    gemm16<<<250, 256, 0, stream>>>(Ah2, Wosw, bout, logits, V_ / 16, KT2, V_);
    softmax_sample<<<B_, 1024, 0, stream>>>(logits, out_s, out_p, sprev, t);
  }
}
